// ContrastiveLossWithDiffusion_317827579955
// MI455X (gfx1250) — compile-verified, matrix operand order checked by reading, not run on hardware
//
#include <hip/hip_runtime.h>
#include <hip/hip_bf16.h>
#include <cstdint>

// ---------------- problem constants (match reference) ----------------
constexpr int NODES = 8192;     // N
constexpr int CH    = 128;      // C
constexpr int EDGES = 262144;   // E
constexpr int STEPS = 10;
constexpr float INV_T = 2.0f;   // 1/TEMPERATURE

typedef __attribute__((ext_vector_type(16))) __bf16   v16bf;
typedef __attribute__((ext_vector_type(8)))  __bf16   v8bf;
typedef __attribute__((ext_vector_type(8)))  float    v8f;
typedef __attribute__((ext_vector_type(4)))  uint32_t v4u;
typedef __attribute__((ext_vector_type(8)))  uint32_t v8u;

union Frag { v16bf v; __bf16 e[16]; };

// ---------------- workspace layout (bytes) ----------------
constexpr size_t DIFF_BYTES = (size_t)NODES * NODES * 2;       // Diff bf16 [N][N]
constexpr size_t DINV_OFF   = DIFF_BYTES;                      // dinv f32 [N]
constexpr size_t Y_OFF      = DINV_OFF + (size_t)NODES * 4;    // Y f32 [N][C]
constexpr size_t XT0_OFF    = Y_OFF + (size_t)NODES * CH * 4;  // Xt0 bf16 [C][N]
constexpr size_t XT1_OFF    = XT0_OFF + (size_t)NODES * CH * 2;// Xt1 bf16 [C][N]
constexpr size_t ACC_OFF    = XT1_OFF + (size_t)NODES * CH * 2;// acc f32 [2]

// ---------------- kernels ----------------

__global__ __launch_bounds__(256) void k_zero_u4(uint4* __restrict__ p, size_t n16) {
    size_t i = (size_t)blockIdx.x * blockDim.x + threadIdx.x;
    size_t stride = (size_t)gridDim.x * blockDim.x;
    uint4 z; z.x = 0u; z.y = 0u; z.z = 0u; z.w = 0u;
    for (; i < n16; i += stride) p[i] = z;
}

__global__ void k_zero_acc(float* __restrict__ acc) {
    if (threadIdx.x < 2) acc[threadIdx.x] = 0.0f;
}

// scatter edges into dense bf16 adjacency (duplicates -> 1, matches .set(1.0))
__global__ __launch_bounds__(256) void k_scatter(const int* __restrict__ ei,
                                                 __bf16* __restrict__ Diff) {
    int e = blockIdx.x * blockDim.x + threadIdx.x;
    if (e < EDGES) {
        int s = ei[e];
        int d = ei[EDGES + e];
        Diff[(size_t)s * NODES + d] = (__bf16)1.0f;
    }
}

// per-row degree -> d^-1/2  (one wave per row; self-loops guarantee degree>=1)
__global__ __launch_bounds__(32) void k_row_degree(const __bf16* __restrict__ Diff,
                                                   float* __restrict__ dinv) {
    int row  = blockIdx.x;
    int lane = threadIdx.x;
    const __bf16* p = Diff + (size_t)row * NODES;
    float s = 0.0f;
    for (int j = lane * 8; j < NODES; j += 32 * 8) {
        v8bf v = *(const v8bf*)(p + j);
        #pragma unroll
        for (int i = 0; i < 8; ++i) s += (float)v[i];
    }
    #pragma unroll
    for (int off = 16; off; off >>= 1) s += __shfl_xor(s, off, 32);
    if (lane == 0) dinv[row] = rsqrtf(s);
}

// Diff[i][j] *= dinv[i]*dinv[j], in place (bf16)
__global__ __launch_bounds__(256) void k_normalize(__bf16* __restrict__ Diff,
                                                   const float* __restrict__ dinv) {
    int row = blockIdx.x;
    float di = dinv[row];
    __bf16* p = Diff + (size_t)row * NODES;
    for (int j = threadIdx.x * 8; j < NODES; j += 256 * 8) {
        v8bf v = *(const v8bf*)(p + j);
        v8bf o;
        #pragma unroll
        for (int i = 0; i < 8; ++i)
            o[i] = (__bf16)((float)v[i] * di * dinv[j + i]);
        *(v8bf*)(p + j) = o;
    }
}

// f32 row-major embeddings -> bf16 transposed operand Xt[CH][NODES]
__global__ __launch_bounds__(256) void k_cvt_xt(const float* __restrict__ emb,
                                                __bf16* __restrict__ Xt) {
    int idx = blockIdx.x * blockDim.x + threadIdx.x;   // over NODES*CH
    if (idx < NODES * CH) {
        int nrow = idx >> 7;          // / CH
        int ch   = idx & (CH - 1);    // % CH
        Xt[(size_t)ch * NODES + nrow] = (__bf16)emb[idx];
    }
}

// ---- Tensor Data Mover: stage Xt[0..127][k0..k0+31] (8 KB slab) into LDS ----
// 2D D# (groups 0+1 only): data_size=2B, tile 32x128, tensor_dim0=NODES,
// tensor_dim1=CH, dim0_stride=NODES, type=2. DMA packs rows at 64 B pitch.
__device__ __forceinline__ void tdm_load_slab(const __bf16* gsrc, uint32_t lds_byte_addr) {
    uint64_t ga = (uint64_t)(uintptr_t)gsrc;
    v4u g0;
    g0[0] = 1u;                                     // count=1, is_restore=0, gather=0
    g0[1] = lds_byte_addr;                          // lds_addr[31:0]
    g0[2] = (uint32_t)ga;                           // global_addr[31:0]
    g0[3] = (uint32_t)(ga >> 32) | (2u << 30);      // global_addr[56:32] | type=2
    v8u g1;
    g1[0] = 1u << 16;                               // workgroup_mask=0, data_size=1 (2B)
    g1[1] = ((uint32_t)NODES & 0xFFFFu) << 16;      // atomic_barrier=0, tensor_dim0 lo16
    g1[2] = ((uint32_t)NODES >> 16) | (((uint32_t)CH & 0xFFFFu) << 16); // dim0 hi16 | tensor_dim1 lo16
    g1[3] = ((uint32_t)CH >> 16) | (32u << 16);     // tensor_dim1 hi16 | tile_dim0=32
    g1[4] = (uint32_t)CH;                           // tile_dim1=128, tile_dim2=0
    g1[5] = (uint32_t)NODES;                        // tensor_dim0_stride lo32
    g1[6] = 0u;                                     // stride hi16 | dim1_stride lo16
    g1[7] = 0u;
    asm volatile("tensor_load_to_lds %0, %1" :: "s"(g0), "s"(g1) : "memory");
}

// One diffusion step: Y[N][C] = Diff(bf16) @ X(bf16);  also emit bf16 Xt_out.
// One wave computes a 16x128 strip (tileM x 8 N-tiles); A streamed from global,
// B staged per-block into LDS by the TDM (double buffered, overlapped with WMMA).
__global__ __launch_bounds__(128) void k_diffuse_gemm(const __bf16* __restrict__ Diff,
                                                      const __bf16* __restrict__ Xt_in,
                                                      __bf16* __restrict__ Xt_out,
                                                      float* __restrict__ Y) {
    __shared__ __align__(16) __bf16 smem[2][CH * 32];   // 2 x 8 KB slabs

    const int lane  = threadIdx.x;                  // 0..31
    const int tileM = blockIdx.x * 4 + threadIdx.y; // 0..511
    const int m     = tileM * 16 + (lane & 15);
    const int kSel  = (lane < 16) ? 0 : 8;          // ISA 16-bit A/B half-wave k split
    const int nlo   = lane & 15;
    const bool loader = (threadIdx.y == 0);

    const uint32_t ldsAddr[2] = { (uint32_t)(uintptr_t)&smem[0][0],
                                  (uint32_t)(uintptr_t)&smem[1][0] };

    v8f acc[8] = {};
    const __bf16* arow = Diff + (size_t)m * NODES;

    // prologue: stage slab 0
    if (loader) {
        tdm_load_slab(Xt_in, ldsAddr[0]);
        __builtin_amdgcn_s_wait_tensorcnt(0);
    }
    __syncthreads();

    for (int k0 = 0; k0 < NODES; k0 += 32) {
        const int cur = (k0 >> 5) & 1;
        // kick off DMA for the next slab into the other buffer
        if (loader && (k0 + 32 < NODES))
            tdm_load_slab(Xt_in + (k0 + 32), ldsAddr[cur ^ 1]);

        const int kb = k0 + kSel;
        if (k0 + 32 < NODES)
            __builtin_prefetch(arow + k0 + 32, 0, 1);   // prefetch next A chunk

        // A fragment from global: elems 0..7 = K kb..kb+7, 8..15 = K kb+16..kb+23
        Frag a;
        {
            const v8bf* ap = (const v8bf*)(arow + kb);
            v8bf lo = ap[0];
            v8bf hi = ap[2];
            #pragma unroll
            for (int i = 0; i < 8; ++i) { a.e[i] = lo[i]; a.e[8 + i] = hi[i]; }
        }

        // all 8 B fragments from LDS slab (row pitch 32 elems = 64 B)
        Frag b[8];
        #pragma unroll
        for (int t = 0; t < 8; ++t) {
            const __bf16* srow = &smem[cur][(size_t)(t * 16 + nlo) * 32 + kSel];
            v8bf lo = *(const v8bf*)(srow);
            v8bf hi = *(const v8bf*)(srow + 16);
            #pragma unroll
            for (int i = 0; i < 8; ++i) { b[t].e[i] = lo[i]; b[t].e[8 + i] = hi[i]; }
        }

        #pragma unroll
        for (int t = 0; t < 8; ++t)
            acc[t] = __builtin_amdgcn_wmma_f32_16x16x32_bf16(
                false, a.v, false, b[t].v, (short)0, acc[t], false, false);

        if (loader) __builtin_amdgcn_s_wait_tensorcnt(0);
        __syncthreads();
    }

    // C/D layout: VGPR r -> M = r (lanes 0..15) / r+8 (lanes 16..31); N = lane&15
    const int rbase = tileM * 16 + ((lane < 16) ? 0 : 8);
    #pragma unroll
    for (int t = 0; t < 8; ++t) {
        const int col = t * 16 + nlo;
        #pragma unroll
        for (int r = 0; r < 8; ++r) {
            const int row = rbase + r;
            const float v = acc[t][r];
            Y[(size_t)row * CH + col]         = v;
            Xt_out[(size_t)col * NODES + row] = (__bf16)v;
        }
    }
}

__device__ __forceinline__ float softplus_stable(float z) {
    return fmaxf(z, 0.0f) + log1pf(expf(-fabsf(z)));
}

// pos loss: one wave per edge; mean of softplus(-dot/T)
__global__ __launch_bounds__(256) void k_pos_loss(const float* __restrict__ Y,
                                                  const int* __restrict__ ei,
                                                  float* __restrict__ acc) {
    int e = blockIdx.x * 8 + threadIdx.y;
    int lane = threadIdx.x;
    int s = ei[e];
    int d = ei[EDGES + e];
    float4 a = *(const float4*)(Y + (size_t)s * CH + lane * 4);
    float4 b = *(const float4*)(Y + (size_t)d * CH + lane * 4);
    float t = a.x * b.x + a.y * b.y + a.z * b.z + a.w * b.w;
    #pragma unroll
    for (int off = 16; off; off >>= 1) t += __shfl_xor(t, off, 32);
    if (lane == 0) atomicAdd(&acc[0], softplus_stable(-(t * INV_T)));
}

// neg loss: one wave per node; mean of softplus(+dot/T)
__global__ __launch_bounds__(256) void k_neg_loss(const float* __restrict__ Y,
                                                  const int* __restrict__ ridx,
                                                  float* __restrict__ acc) {
    int n = blockIdx.x * 8 + threadIdx.y;
    int lane = threadIdx.x;
    int r = ridx[n];
    float4 a = *(const float4*)(Y + (size_t)n * CH + lane * 4);
    float4 b = *(const float4*)(Y + (size_t)r * CH + lane * 4);
    float t = a.x * b.x + a.y * b.y + a.z * b.z + a.w * b.w;
    #pragma unroll
    for (int off = 16; off; off >>= 1) t += __shfl_xor(t, off, 32);
    if (lane == 0) atomicAdd(&acc[1], softplus_stable(t * INV_T));
}

__global__ void k_finalize(const float* __restrict__ acc, float* __restrict__ out) {
    out[0] = acc[0] * (1.0f / (float)EDGES) + acc[1] * (1.0f / (float)NODES);
}

// ---------------- launch ----------------
extern "C" void kernel_launch(void* const* d_in, const int* in_sizes, int n_in,
                              void* d_out, int out_size, void* d_ws, size_t ws_size,
                              hipStream_t stream) {
    const float* emb  = (const float*)d_in[0];   // [NODES][CH] f32
    const int*   ei   = (const int*)d_in[1];     // [2][EDGES]  int32
    const int*   ridx = (const int*)d_in[2];     // [NODES]     int32
    float* out = (float*)d_out;

    char* ws = (char*)d_ws;
    __bf16* Diff = (__bf16*)ws;
    float*  dinv = (float*)(ws + DINV_OFF);
    float*  Yf   = (float*)(ws + Y_OFF);
    __bf16* Xt0  = (__bf16*)(ws + XT0_OFF);
    __bf16* Xt1  = (__bf16*)(ws + XT1_OFF);
    float*  acc  = (float*)(ws + ACC_OFF);

    // 1) zero adjacency + accumulators (must happen every call: in-place normalize)
    k_zero_u4<<<2048, 256, 0, stream>>>((uint4*)Diff, DIFF_BYTES / 16);
    k_zero_acc<<<1, 32, 0, stream>>>(acc);

    // 2) scatter edges (duplicates -> 1)
    k_scatter<<<(EDGES + 255) / 256, 256, 0, stream>>>(ei, Diff);

    // 3) degree^-1/2 per row
    k_row_degree<<<NODES, 32, 0, stream>>>(Diff, dinv);

    // 4) symmetric normalization in place (bf16)
    k_normalize<<<NODES, 256, 0, stream>>>(Diff, dinv);

    // 5) embeddings -> bf16 transposed operand
    k_cvt_xt<<<(NODES * CH + 255) / 256, 256, 0, stream>>>(emb, Xt0);

    // 6) 10 diffusion steps: Y = Diff @ X (WMMA bf16 -> f32), TDM-staged operand
    for (int s = 0; s < STEPS; ++s) {
        __bf16* xin  = (s & 1) ? Xt1 : Xt0;
        __bf16* xout = (s & 1) ? Xt0 : Xt1;
        k_diffuse_gemm<<<dim3(NODES / 16 / 4), dim3(32, 4), 0, stream>>>(Diff, xin, xout, Yf);
    }

    // 7) contrastive losses on final f32 embeddings
    k_pos_loss<<<EDGES / 8, dim3(32, 8), 0, stream>>>(Yf, ei, acc);
    k_neg_loss<<<NODES / 8, dim3(32, 8), 0, stream>>>(Yf, ridx, acc);

    // 8) scalar output
    k_finalize<<<1, 1, 0, stream>>>(acc, out);
}
